// MoELayer_4612794876348
// MI455X (gfx1250) — matrix operand order repaired and checked by reading, not compiled, not run on hardware
//
#include <hip/hip_runtime.h>
#include <math.h>

// ---------------------------------------------------------------------------
// MoE layer for MI455X (gfx1250, wave32, WMMA + TDM).
// Compute-bound (~96 GFLOP vs ~100 MB traffic, fits in 192MB L2): all dense
// contractions run through v_wmma_f32_16x16x32_f16.  Weights are converted to
// f16 once per launch; B tiles are staged into LDS by the Tensor Data Mover
// (tensor_load_to_lds + s_wait_tensorcnt) with hardware padding, falling back
// to vectorized b128 copies on partial tiles / toolchains without the builtin.
// ---------------------------------------------------------------------------

#define B_    16
#define C_    384
#define HW_   1024
#define NT_   16384            // B_*HW_ token rows
#define DH_   48
#define NH_   8
#define ACT_NONE 0
#define ACT_GELU 1
#define ACT_SIGM 2
#define BN_SCL 0.9999950000374997f   // 1/sqrt(1+1e-5)

typedef __attribute__((ext_vector_type(16))) _Float16 v16h;
typedef __attribute__((ext_vector_type(8)))  float    v8f;
typedef __attribute__((ext_vector_type(4)))  _Float16 h4;
typedef __attribute__((ext_vector_type(8)))  _Float16 h8;
typedef __attribute__((ext_vector_type(4)))  float    f4;
typedef __attribute__((ext_vector_type(4)))  unsigned int u32x4;
typedef __attribute__((ext_vector_type(8)))  int      i32x8;
typedef __attribute__((ext_vector_type(4)))  int      i32x4;

#if defined(__gfx1250__) && __has_builtin(__builtin_amdgcn_tensor_load_to_lds) && \
    __has_builtin(__builtin_amdgcn_s_wait_tensorcnt)
#define HAVE_TDM 1
#endif

__device__ __forceinline__ float gelu_exact(float x) {
  return 0.5f * x * (1.0f + erff(x * 0.70710678118654752f));
}

// token row -> shifted token row (zero-pad boundaries), n = b*1024 + h*32 + w
__device__ __forceinline__ int tok_shift(int n, int dy, int dx) {
  int b = n >> 10, hw = n & 1023, h = hw >> 5, w = hw & 31;
  h += dy; w += dx;
  if (h < 0 || h > 31 || w < 0 || w > 31) return -1;
  return (b << 10) + (h << 5) + w;
}

// ---- WMMA fragment builders (wave32, contiguous k = 16*(lane/16) + s in A and B) -------------------
__device__ __forceinline__ v16h frag_a(const _Float16* p, int ld) {
  int lane = threadIdx.x & 31;
  int r = lane & 15;
  int kg = (lane >> 4) << 4;
  v16h f;
#pragma unroll
  for (int j = 0; j < 8; ++j) {
    int kb = (j << 1) + kg;
    f[2 * j]     = p[r * ld + kb];
    f[2 * j + 1] = p[r * ld + kb + 1];
  }
  return f;
}
__device__ __forceinline__ v16h frag_b_nk(const _Float16* p, int ld) {
  int lane = threadIdx.x & 31;
  int n = lane & 15;
  int kg = (lane >> 4) << 4;
  v16h f;
#pragma unroll
  for (int i = 0; i < 16; ++i) f[i] = p[n * ld + kg + i];
  return f;
}
__device__ __forceinline__ v16h frag_b_kn(const _Float16* p, int ld) {
  int lane = threadIdx.x & 31;
  int n = lane & 15;
  int kg = (lane >> 4) << 4;
  v16h f;
#pragma unroll
  for (int i = 0; i < 16; ++i) f[i] = p[(kg + i) * ld + n];
  return f;
}

// ---------------------------------------------------------------------------
// Token-major GEMM, f16 weights:  Y[m, ycol0+o] (+= if accum) =
//   epilogue( sum_k X[shift(m), k] * W[o*K + k] )
// epilogue: +bias -> (+prev if accum) -> BN fold -> act -> +res
// Block 128(M) x 64(O), 256 threads = 8 waves, wave w owns rows w*16..+15.
// ---------------------------------------------------------------------------
__global__ __launch_bounds__(256)
void gemm_tok(const float* __restrict__ X, int xld,
              const _Float16* __restrict__ W,
              const float* __restrict__ bias,
              const float* __restrict__ scale, const float* __restrict__ shift,
              const float* __restrict__ res,
              float* __restrict__ Y, int yld, int ycol0,
              int M, int K, int O, int dy, int dx, int accum, int act) {
  __shared__ _Float16 la[128][40];
  __shared__ _Float16 lb[64][40];
  const int tid = threadIdx.x;
  const int wave = tid >> 5, lane = tid & 31;
  const int bm = blockIdx.y * 128;
  const int bo = blockIdx.x * 64;
  const bool shifted = (dy != 0) || (dx != 0);

  v8f acc0 = {}, acc1 = {}, acc2 = {}, acc3 = {};

  for (int k0 = 0; k0 < K; k0 += 32) {
    // ---- stage A tile (f32 -> f16), float4 per thread-trip -----------------
    for (int t = tid; t < 128 * 8; t += 256) {
      int i = t >> 3, kc = (t & 7) << 2;
      int m = bm + i;
      int src = (m < M) ? (shifted ? tok_shift(m, dy, dx) : m) : -1;
      f4 v = {0.f, 0.f, 0.f, 0.f};
      if (src >= 0 && (k0 + kc) < K)
        v = *(const f4*)(X + (size_t)src * xld + k0 + kc);
      h4 hv;
      hv[0] = (_Float16)v[0]; hv[1] = (_Float16)v[1];
      hv[2] = (_Float16)v[2]; hv[3] = (_Float16)v[3];
      *(h4*)(&la[i][kc]) = hv;
    }
    if (k0 + 32 < K)   // CDNA5 global_prefetch_b8 of next A chunk
      __builtin_prefetch(X + (size_t)(bm + (tid >> 1)) * xld + k0 + 32, 0, 1);

    // ---- stage B tile (f16 weights) ----------------------------------------
    const bool fullB = (bo + 64 <= O) && (k0 + 32 <= K);
#ifdef HAVE_TDM
    if (fullB) {
      if (wave == 0) {
        // Tensor DMA: 32(K) x 64(O) f16 tile, row-major [O,K] source,
        // LDS padding 4 DWORDs after every 16 DWORDs -> row pitch 40 halfs.
        unsigned long long ga =
            (unsigned long long)(const void*)(W + (size_t)bo * K + k0);
        unsigned int lds_addr = (unsigned int)(size_t)(&lb[0][0]);
        u32x4 g0 = { 1u,                                     // count=1 user D#
                     lds_addr,
                     (unsigned int)(ga & 0xFFFFFFFFu),
                     (unsigned int)((ga >> 32) & 0x01FFFFFFu) | (2u << 30) };
        unsigned int g1w0 = (1u << 16)      // data_size = 2 bytes
                          | (1u << 20)      // pad_enable
                          | (3u << 22)      // pad_interval: 16 DWORDs
                          | (3u << 25);     // pad_amount: 4 DWORDs
        i32x8 g1 = { (int)g1w0,
                     (int)(((unsigned)K & 0xFFFFu) << 16),                 // tdim0 lo
                     (int)(((unsigned)K >> 16) | (((unsigned)O & 0xFFFFu) << 16)),
                     (int)(((unsigned)O >> 16) | (32u << 16)),             // tile_dim0=32
                     64,                                                   // tile_dim1=64
                     K,                                                    // dim0 stride
                     0, 0 };
        i32x4 z4 = {0, 0, 0, 0};
        i32x8 z8 = {0, 0, 0, 0, 0, 0, 0, 0};
        __builtin_amdgcn_tensor_load_to_lds(g0, g1, z4, z4, z8, 0);
        __builtin_amdgcn_s_wait_tensorcnt(0);
      }
    } else
#endif
    {
      for (int t = tid; t < 64 * 4; t += 256) {   // h8 per thread-trip
        int o = t >> 2, kc = (t & 3) << 3;
        int oo = bo + o;
        h8 v = {};
        if (fullB || (oo < O && (k0 + kc) < K))
          v = *(const h8*)(W + (size_t)oo * K + k0 + kc);
        *(h8*)(&lb[o][kc]) = v;
      }
    }
    __syncthreads();

    v16h a = frag_a(&la[wave * 16][0], 40);
    {
      v16h b0 = frag_b_nk(&lb[0][0], 40);
      acc0 = __builtin_amdgcn_wmma_f32_16x16x32_f16(false, a, false, b0, (short)0, acc0, false, false);
      v16h b1 = frag_b_nk(&lb[16][0], 40);
      acc1 = __builtin_amdgcn_wmma_f32_16x16x32_f16(false, a, false, b1, (short)0, acc1, false, false);
      v16h b2 = frag_b_nk(&lb[32][0], 40);
      acc2 = __builtin_amdgcn_wmma_f32_16x16x32_f16(false, a, false, b2, (short)0, acc2, false, false);
      v16h b3 = frag_b_nk(&lb[48][0], 40);
      acc3 = __builtin_amdgcn_wmma_f32_16x16x32_f16(false, a, false, b3, (short)0, acc3, false, false);
    }
    __syncthreads();
  }

  // epilogue: D layout -> lane holds col n=lane&15, vgpr r -> row r (+8 hi)
  const int n15 = lane & 15;
  const int hi = (lane & 16) ? 8 : 0;
  v8f accs[4] = {acc0, acc1, acc2, acc3};
#pragma unroll
  for (int t4 = 0; t4 < 4; ++t4) {
    int col = bo + t4 * 16 + n15;
    if (col >= O) continue;
#pragma unroll
    for (int r = 0; r < 8; ++r) {
      int m = bm + wave * 16 + r + hi;
      if (m >= M) continue;
      float v = accs[t4][r];
      if (bias)  v += bias[col];
      float* yp = &Y[(size_t)m * yld + ycol0 + col];
      if (accum) v += *yp;
      if (scale) v = v * scale[col] * BN_SCL + shift[col];
      if (act == ACT_GELU) v = gelu_exact(v);
      else if (act == ACT_SIGM) v = 1.f / (1.f + __expf(-v));
      if (res) v += res[(size_t)m * C_ + col];
      *yp = v;
    }
  }
}

// ---------------------------------------------------------------------------
// Flash-style MHSA: block = (b, head, 128-query slab); 8 waves x 16 queries.
// 32-key tiles; QK^T with dh=48 zero-padded to 64 (2 chunks); online softmax;
// P routed through per-wave LDS back into A-fragment form for P*V.
// ---------------------------------------------------------------------------
__global__ __launch_bounds__(256)
void attn_kernel(const float* __restrict__ QKV, float* __restrict__ O) {
  const int blk  = blockIdx.x;
  const int qblk = blk & 7;
  const int head = (blk >> 3) & 7;
  const int b    = blk >> 6;
  const int tid  = threadIdx.x;
  const int wave = tid >> 5, lane = tid & 31;
  const int qrow0 = qblk * 128 + wave * 16;

  const size_t base = (size_t)b * HW_ * (3 * C_);
  const float* Qp = QKV + base + 0 * C_ + head * DH_;
  const float* Kp = QKV + base + 1 * C_ + head * DH_;
  const float* Vp = QKV + base + 2 * C_ + head * DH_;

  __shared__ _Float16 kl[32][72];
  __shared__ _Float16 vl[32][72];
  __shared__ _Float16 pl[8][16][40];

  v16h aq0, aq1;
  {
    int r = lane & 15;
    int kg = (lane >> 4) << 4;
    const float* qr = Qp + (size_t)(qrow0 + r) * (3 * C_);
#pragma unroll
    for (int j = 0; j < 8; ++j) {
      int kb = (j << 1) + kg;
      aq0[2 * j]     = (_Float16)qr[kb];
      aq0[2 * j + 1] = (_Float16)qr[kb + 1];
      int d1 = 32 + kb;
      aq1[2 * j]     = (d1     < DH_) ? (_Float16)qr[d1]     : (_Float16)0.f;
      aq1[2 * j + 1] = (d1 + 1 < DH_) ? (_Float16)qr[d1 + 1] : (_Float16)0.f;
    }
  }

  float mrow[8], lrow[8];
  v8f o0 = {}, o1 = {}, o2 = {};
#pragma unroll
  for (int r = 0; r < 8; ++r) { mrow[r] = -3.0e38f; lrow[r] = 0.f; }
  const float sc = 0.14433756729740643f;  // 1/sqrt(48)

  for (int kt = 0; kt < 32; ++kt) {
    const int krow0 = kt * 32;
    // vectorized K/V staging: float4 per trip, zero-pad d>=48
    for (int t = tid; t < 32 * 16; t += 256) {
      int kk = t >> 4, dc = (t & 15) << 2;
      f4 kv = {0.f, 0.f, 0.f, 0.f}, vv = {0.f, 0.f, 0.f, 0.f};
      if (dc < DH_) {
        kv = *(const f4*)(Kp + (size_t)(krow0 + kk) * (3 * C_) + dc);
        vv = *(const f4*)(Vp + (size_t)(krow0 + kk) * (3 * C_) + dc);
      }
      h4 kh, vh;
      kh[0] = (_Float16)kv[0]; kh[1] = (_Float16)kv[1];
      kh[2] = (_Float16)kv[2]; kh[3] = (_Float16)kv[3];
      vh[0] = (_Float16)vv[0]; vh[1] = (_Float16)vv[1];
      vh[2] = (_Float16)vv[2]; vh[3] = (_Float16)vv[3];
      *(h4*)(&kl[kk][dc]) = kh;
      *(h4*)(&vl[kk][dc]) = vh;
    }
    __syncthreads();

    v8f s0 = {}, s1 = {};
    {
      v16h b0 = frag_b_nk(&kl[0][0], 72);
      s0 = __builtin_amdgcn_wmma_f32_16x16x32_f16(false, aq0, false, b0, (short)0, s0, false, false);
      v16h b1 = frag_b_nk(&kl[16][0], 72);
      s1 = __builtin_amdgcn_wmma_f32_16x16x32_f16(false, aq0, false, b1, (short)0, s1, false, false);
      v16h b2 = frag_b_nk(&kl[0][32], 72);
      s0 = __builtin_amdgcn_wmma_f32_16x16x32_f16(false, aq1, false, b2, (short)0, s0, false, false);
      v16h b3 = frag_b_nk(&kl[16][32], 72);
      s1 = __builtin_amdgcn_wmma_f32_16x16x32_f16(false, aq1, false, b3, (short)0, s1, false, false);
    }

#pragma unroll
    for (int r = 0; r < 8; ++r) { s0[r] *= sc; s1[r] *= sc; }
    float alpha[8];
#pragma unroll
    for (int r = 0; r < 8; ++r) {
      float m = fmaxf(s0[r], s1[r]);
#pragma unroll
      for (int off = 1; off < 16; off <<= 1) m = fmaxf(m, __shfl_xor(m, off, 32));
      float mnew = fmaxf(mrow[r], m);
      alpha[r] = __expf(mrow[r] - mnew);
      mrow[r] = mnew;
      s0[r] = __expf(s0[r] - mnew);
      s1[r] = __expf(s1[r] - mnew);
      float rs = s0[r] + s1[r];
#pragma unroll
      for (int off = 1; off < 16; off <<= 1) rs += __shfl_xor(rs, off, 32);
      lrow[r] = lrow[r] * alpha[r] + rs;
      o0[r] *= alpha[r]; o1[r] *= alpha[r]; o2[r] *= alpha[r];
    }
    {
      int n = lane & 15, hi = (lane >> 4) & 1;
#pragma unroll
      for (int r = 0; r < 8; ++r) {
        pl[wave][r + 8 * hi][n]      = (_Float16)s0[r];
        pl[wave][r + 8 * hi][16 + n] = (_Float16)s1[r];
      }
    }
    __syncthreads();
    {
      v16h ap  = frag_a(&pl[wave][0][0], 40);
      v16h bv0 = frag_b_kn(&vl[0][0], 72);
      o0 = __builtin_amdgcn_wmma_f32_16x16x32_f16(false, ap, false, bv0, (short)0, o0, false, false);
      v16h bv1 = frag_b_kn(&vl[0][16], 72);
      o1 = __builtin_amdgcn_wmma_f32_16x16x32_f16(false, ap, false, bv1, (short)0, o1, false, false);
      v16h bv2 = frag_b_kn(&vl[0][32], 72);
      o2 = __builtin_amdgcn_wmma_f32_16x16x32_f16(false, ap, false, bv2, (short)0, o2, false, false);
    }
    __syncthreads();
  }

  {
    int n = lane & 15, hi = (lane >> 4) & 1;
#pragma unroll
    for (int r = 0; r < 8; ++r) {
      size_t row = (size_t)b * HW_ + qrow0 + r + 8 * hi;
      float inv = 1.f / lrow[r];
      float* op = O + row * C_ + head * DH_;
      op[n]      = o0[r] * inv;
      op[16 + n] = o1[r] * inv;
      op[32 + n] = o2[r] * inv;
    }
  }
}

// ---------------------------------------------------------------------------
// VALU kernels
// ---------------------------------------------------------------------------
__global__ void cvt16_k(const float* __restrict__ X, _Float16* __restrict__ Y, int n) {
  int i = blockIdx.x * 256 + threadIdx.x;
  if (i < n) Y[i] = (_Float16)X[i];
}
// wk [96,96,3,3] -> 9 contiguous f16 [96,96] tap matrices: Y[tap][o][c]
__global__ void cvt_tap_k(const float* __restrict__ wk, _Float16* __restrict__ Y) {
  int i = blockIdx.x * 256 + threadIdx.x;
  if (i >= 9 * 96 * 96) return;
  int tap = i / (96 * 96), rem = i % (96 * 96), o = rem / 96, c = rem % 96;
  Y[i] = (_Float16)wk[(size_t)(o * 96 + c) * 9 + tap];
}
__global__ void nchw2tok_k(const float* __restrict__ X, float* __restrict__ Y) {
  int i = blockIdx.x * 256 + threadIdx.x;
  if (i >= NT_ * C_) return;
  int n = i / C_, c = i % C_;
  int b = n >> 10, hw = n & 1023;
  Y[i] = X[((size_t)b * C_ + c) * HW_ + hw];
}
__global__ void tok2nchw_k(const float* __restrict__ X, float* __restrict__ Y) {
  int i = blockIdx.x * 256 + threadIdx.x;
  if (i >= NT_ * C_) return;
  int n = i / C_, c = i % C_;
  int b = n >> 10, hw = n & 1023;
  Y[((size_t)b * C_ + c) * HW_ + hw] = X[i];
}
__global__ void zero_k(float* __restrict__ p, int n) {
  int i = blockIdx.x * 256 + threadIdx.x;
  if (i < n) p[i] = 0.f;
}
__global__ __launch_bounds__(128)
void ln_k(const float* __restrict__ X, const float* __restrict__ G,
          const float* __restrict__ Bb, const float* __restrict__ RES,
          float* __restrict__ Y, int act) {
  __shared__ float red[128];
  const int n = blockIdx.x, tid = threadIdx.x;
  float s = 0.f, ss = 0.f;
  for (int c = tid; c < C_; c += 128) {
    float v = X[(size_t)n * C_ + c];
    s += v; ss += v * v;
  }
  red[tid] = s; __syncthreads();
  for (int o = 64; o; o >>= 1) { if (tid < o) red[tid] += red[tid + o]; __syncthreads(); }
  float mean = red[0] * (1.f / C_); __syncthreads();
  red[tid] = ss; __syncthreads();
  for (int o = 64; o; o >>= 1) { if (tid < o) red[tid] += red[tid + o]; __syncthreads(); }
  float var = red[0] * (1.f / C_) - mean * mean;
  float inv = rsqrtf(var + 1e-5f);
  for (int c = tid; c < C_; c += 128) {
    float v = (X[(size_t)n * C_ + c] - mean) * inv * G[c] + Bb[c];
    if (act) v = gelu_exact(v);
    if (RES) v += RES[(size_t)n * C_ + c];
    Y[(size_t)n * C_ + c] = v;
  }
}
__global__ void gap_k(const float* __restrict__ X, float* __restrict__ G) {
  int idx = blockIdx.x * 256 + threadIdx.x;
  if (idx >= B_ * C_) return;
  int b = idx / C_, c = idx % C_;
  const float* p = X + (size_t)b * HW_ * C_ + c;
  float s = 0.f;
  for (int i = 0; i < HW_; ++i) s += p[(size_t)i * C_];
  G[idx] = s * (1.f / HW_);
}
__global__ void bmul_k(const float* __restrict__ X, const float* __restrict__ Gt,
                       float* __restrict__ Y) {
  int i = blockIdx.x * 256 + threadIdx.x;
  if (i >= NT_ * C_) return;
  int n = i / C_, c = i % C_;
  Y[i] = X[i] * Gt[(n >> 10) * C_ + c];
}
__global__ void combine_k(float* __restrict__ ACC, const float* __restrict__ S,
                          const float* __restrict__ CW, int e) {
  int i = blockIdx.x * 256 + threadIdx.x;
  if (i >= NT_ * C_) return;
  int n = i / C_;
  ACC[i] += CW[(n >> 10) * 5 + e] * S[i];
}
__global__ void dw5_k(const float* __restrict__ X, const float* __restrict__ Wt,
                      const float* __restrict__ Bi, float* __restrict__ Y) {
  int i = blockIdx.x * 256 + threadIdx.x;
  if (i >= NT_ * C_) return;
  int n = i / C_, c = i % C_;
  float s = Bi[c];
#pragma unroll
  for (int ki = 0; ki < 5; ++ki)
#pragma unroll
    for (int kj = 0; kj < 5; ++kj) {
      int src = tok_shift(n, ki - 2, kj - 2);
      if (src >= 0) s += X[(size_t)src * C_ + c] * Wt[c * 25 + ki * 5 + kj];
    }
  Y[i] = s;
}
__global__ void edge_k(const float* __restrict__ X, float* __restrict__ SB,
                       float* __restrict__ LP, float* __restrict__ DG,
                       float* __restrict__ GR) {
  int i = blockIdx.x * 256 + threadIdx.x;
  if (i >= NT_ * C_) return;
  int n = i / C_, c = i % C_;
  float v[3][3];
#pragma unroll
  for (int a = 0; a < 3; ++a)
#pragma unroll
    for (int bb = 0; bb < 3; ++bb) {
      int src = tok_shift(n, a - 1, bb - 1);
      v[a][bb] = (src >= 0) ? X[(size_t)src * C_ + c] : 0.f;
    }
  float sx = -v[0][0] + v[0][2] - 2.f * v[1][0] + 2.f * v[1][2] - v[2][0] + v[2][2];
  float sy = -v[0][0] - 2.f * v[0][1] - v[0][2] + v[2][0] + 2.f * v[2][1] + v[2][2];
  float sob = sqrtf(sx * sx + sy * sy + 1e-8f);
  float lap = fabsf(v[0][1] + v[1][0] - 4.f * v[1][1] + v[1][2] + v[2][1]);
  float d1 = v[0][0] - v[0][2] - v[2][0] + v[2][2];
  float d2 = -v[0][0] + v[0][2] + v[2][0] - v[2][2];
  float dg = fmaxf(fabsf(d1), fabsf(d2));
  SB[i] = sob; LP[i] = lap; DG[i] = dg;
  GR[i] = sqrtf(sob * sob + lap * lap + 1e-8f);
}
__global__ void freq_k(const float* __restrict__ X, float* __restrict__ LO,
                       float* __restrict__ MI, float* __restrict__ HI) {
  int i = blockIdx.x * 256 + threadIdx.x;
  if (i >= NT_ * C_) return;
  int n = i / C_, c = i % C_;
  float s3 = 0.f, s5 = 0.f;
#pragma unroll
  for (int a = -2; a <= 2; ++a)
#pragma unroll
    for (int bb = -2; bb <= 2; ++bb) {
      int src = tok_shift(n, a, bb);
      float vv = (src >= 0) ? X[(size_t)src * C_ + c] : 0.f;
      s5 += vv;
      if (a >= -1 && a <= 1 && bb >= -1 && bb <= 1) s3 += vv;
    }
  float a3 = s3 * (1.f / 9.f), a5 = s5 * (1.f / 25.f);
  LO[i] = a3;
  HI[i] = X[i] - a3;
  MI[i] = a3 - a5;
}
__global__ __launch_bounds__(256)
void router_k(const float* __restrict__ XT, const float* __restrict__ W1,
              const float* __restrict__ B1, const float* __restrict__ W2,
              const float* __restrict__ B2, float* __restrict__ CW) {
  __shared__ float g[C_], h[96], lg[8];
  const int b = blockIdx.x, tid = threadIdx.x;
  for (int c = tid; c < C_; c += 256) {
    const float* p = XT + (size_t)b * HW_ * C_ + c;
    float s = 0.f;
    for (int i = 0; i < HW_; ++i) s += p[(size_t)i * C_];
    g[c] = s * (1.f / HW_);
  }
  __syncthreads();
  if (tid < 96) {
    float s = B1[tid];
    for (int c = 0; c < C_; ++c) s += g[c] * W1[tid * C_ + c];
    h[tid] = gelu_exact(s);
  }
  __syncthreads();
  if (tid < 5) {
    float s = B2[tid];
    for (int c = 0; c < 96; ++c) s += h[c] * W2[tid * 96 + c];
    lg[tid] = s;
  }
  __syncthreads();
  if (tid == 0) {
    int idx[3]; float val[3];
    bool used[5] = {false, false, false, false, false};
    for (int t = 0; t < 3; ++t) {
      int bi = 0; float bv = -3e38f;
      for (int e = 0; e < 5; ++e)
        if (!used[e] && lg[e] > bv) { bv = lg[e]; bi = e; }
      used[bi] = true; idx[t] = bi; val[t] = bv;
    }
    float den = 0.f, w[3];
    for (int t = 0; t < 3; ++t) { w[t] = __expf(val[t] - val[0]); den += w[t]; }
    for (int e = 0; e < 5; ++e) CW[b * 5 + e] = 0.f;
    for (int t = 0; t < 3; ++t) CW[b * 5 + idx[t]] = w[t] / den;
  }
}

// ---------------------------------------------------------------------------
// host-side orchestration
// ---------------------------------------------------------------------------
static inline void launch_gemm(hipStream_t st, const float* X, int xld,
                               const _Float16* W, const float* bias,
                               const float* scale, const float* shift,
                               const float* res, float* Y, int yld, int ycol0,
                               int M, int K, int O, int dy, int dx, int accum, int act) {
  dim3 g((unsigned)((O + 63) / 64), (unsigned)((M + 127) / 128));
  gemm_tok<<<g, 256, 0, st>>>(X, xld, W, bias, scale, shift, res,
                              Y, yld, ycol0, M, K, O, dy, dx, accum, act);
}

#define EWG ((NT_ * C_ + 255) / 256)

static void run_se_fuse(hipStream_t st, void* const* d_in, int base,
                        const float* f0, const float* f1, const float* f2,
                        const float* f3, const float* XT, float* CAT,
                        float* GAP, float* MID, float* GATE, float* OUT,
                        _Float16** whp) {
  auto cvtw = [&](int idx, int n) -> const _Float16* {
    _Float16* p = *whp;
    cvt16_k<<<(n + 255) / 256, 256, 0, st>>>((const float*)d_in[idx], p, n);
    *whp += (size_t)((n + 15) & ~15);
    return p;
  };
  const float* feats[4] = {f0, f1, f2, f3};
  for (int q = 0; q < 4; ++q) {
    const _Float16* w = cvtw(base + 4 * q, 96 * C_);
    const float* bb = (const float*)d_in[base + 4 * q + 1];
    const float* g  = (const float*)d_in[base + 4 * q + 2];
    const float* be = (const float*)d_in[base + 4 * q + 3];
    launch_gemm(st, feats[q], C_, w, bb, g, be, nullptr,
                CAT, C_, q * 96, NT_, C_, 96, 0, 0, 0, ACT_GELU);
  }
  const _Float16* aw1h = cvtw(base + 16, 48 * C_);
  const float*    ab1  = (const float*)d_in[base + 17];
  const _Float16* aw2h = cvtw(base + 18, C_ * 48);
  const float*    ab2  = (const float*)d_in[base + 19];
  const _Float16* wfh  = cvtw(base + 20, C_ * C_);
  const float*    bf   = (const float*)d_in[base + 21];
  const float*    lg   = (const float*)d_in[base + 22];
  const float*    lb   = (const float*)d_in[base + 23];
  gap_k<<<(B_ * C_ + 255) / 256, 256, 0, st>>>(CAT, GAP);
  launch_gemm(st, GAP, C_, aw1h, ab1, nullptr, nullptr, nullptr,
              MID, 48, 0, B_, C_, 48, 0, 0, 0, ACT_GELU);
  launch_gemm(st, MID, 48, aw2h, ab2, nullptr, nullptr, nullptr,
              GATE, C_, 0, B_, 48, C_, 0, 0, 0, ACT_SIGM);
  bmul_k<<<EWG, 256, 0, st>>>(CAT, GATE, CAT);
  launch_gemm(st, CAT, C_, wfh, bf, nullptr, nullptr, nullptr,
              OUT, C_, 0, NT_, C_, C_, 0, 0, 0, ACT_NONE);
  ln_k<<<NT_, 128, 0, st>>>(OUT, lg, lb, XT, OUT, 1);
}

extern "C" void kernel_launch(void* const* d_in, const int* in_sizes, int n_in,
                              void* d_out, int out_size, void* d_ws, size_t ws_size,
                              hipStream_t stream) {
  (void)in_sizes; (void)n_in; (void)out_size; (void)ws_size;
  // Param flattening assumption (recursive insertion order of setup_inputs()):
  //  0:x  1-4:router  5-20:tex branches  21-24:tex wf,bf,g,b
  //  25-30:att  31-42:hyb  43-66:freq  67-90:edge
  const float* x = (const float*)d_in[0];
#define IN(i) ((const float*)d_in[(i)])

  float* ws = (float*)d_ws;
  const size_t S = (size_t)NT_ * C_;
  float* XT   = ws;
  float* ACC  = ws + 1 * S;
  float* T0   = ws + 2 * S;
  float* T1   = ws + 3 * S;
  float* T2   = ws + 4 * S;
  float* T3   = ws + 5 * S;
  float* CAT  = ws + 6 * S;
  float* QKV  = ws + 7 * S;          // 3*S
  float* SM   = ws + 10 * S;
  float* GAP  = SM;                  // B_*C_
  float* GATE = SM + 8192;
  float* CW   = SM + 16384;
  float* MID  = SM + 17408;
  _Float16* wh = (_Float16*)(SM + 32768);   // f16 weight arena (~4.2 MB)

  auto cvtw = [&](int idx, int n) -> const _Float16* {
    _Float16* p = wh;
    cvt16_k<<<(n + 255) / 256, 256, 0, stream>>>(IN(idx), p, n);
    wh += (size_t)((n + 15) & ~15);
    return p;
  };

  // ---- prep ----
  nchw2tok_k<<<EWG, 256, 0, stream>>>(x, XT);
  zero_k<<<EWG, 256, 0, stream>>>(ACC, NT_ * C_);
  router_k<<<B_, 256, 0, stream>>>(XT, IN(1), IN(2), IN(3), IN(4), CW);

  // ---- expert 0: texture ----
  for (int bi = 0; bi < 4; ++bi) {
    const int d = bi + 1;
    const _Float16* w0h = cvtw(5 + 4 * bi + 0, 96 * C_);
    const float* b0 = IN(5 + 4 * bi + 1);
    const float* bk = IN(5 + 4 * bi + 3);
    _Float16* wkh = wh;
    cvt_tap_k<<<(9 * 96 * 96 + 255) / 256, 256, 0, stream>>>(IN(5 + 4 * bi + 2), wkh);
    wh += 9 * 96 * 96;
    launch_gemm(stream, XT, C_, w0h, b0, nullptr, nullptr, nullptr,
                T0, 96, 0, NT_, C_, 96, 0, 0, 0, ACT_NONE);
    int tap = 0;
    for (int ki = 0; ki < 3; ++ki)
      for (int kj = 0; kj < 3; ++kj, ++tap)
        launch_gemm(stream, T0, 96, wkh + (size_t)tap * 96 * 96,
                    (tap == 0) ? bk : nullptr, nullptr, nullptr, nullptr,
                    CAT, C_, bi * 96, NT_, 96, 96,
                    (ki - 1) * d, (kj - 1) * d, tap > 0,
                    (tap == 8) ? ACT_GELU : ACT_NONE);
  }
  {
    const _Float16* wfh = cvtw(21, C_ * C_);
    launch_gemm(stream, CAT, C_, wfh, IN(22), nullptr, nullptr, nullptr,
                T1, C_, 0, NT_, C_, C_, 0, 0, 0, ACT_NONE);
    ln_k<<<NT_, 128, 0, stream>>>(T1, IN(23), IN(24), XT, T2, 1);
    combine_k<<<EWG, 256, 0, stream>>>(ACC, T2, CW, 0);
  }

  // ---- expert 1: attention ----
  {
    const _Float16* wqkvh = cvtw(27, 3 * C_ * C_);
    const _Float16* woh   = cvtw(29, C_ * C_);
    ln_k<<<NT_, 128, 0, stream>>>(XT, IN(25), IN(26), nullptr, T0, 0);
    launch_gemm(stream, T0, C_, wqkvh, IN(28), nullptr, nullptr, nullptr,
                QKV, 3 * C_, 0, NT_, C_, 3 * C_, 0, 0, 0, ACT_NONE);
    attn_kernel<<<B_ * NH_ * 8, 256, 0, stream>>>(QKV, T1);
    launch_gemm(stream, T1, C_, woh, IN(30), nullptr, nullptr, XT,
                T2, C_, 0, NT_, C_, C_, 0, 0, 0, ACT_NONE);
    combine_k<<<EWG, 256, 0, stream>>>(ACC, T2, CW, 1);
  }

  // ---- expert 2: hybrid ----
  {
    const _Float16* w1h  = cvtw(35, C_ * C_);
    const _Float16* gw1h = cvtw(37, 96 * C_);
    const _Float16* gw2h = cvtw(39, C_ * 96);
    const _Float16* wfh  = cvtw(41, C_ * C_);
    dw5_k<<<EWG, 256, 0, stream>>>(XT, IN(31), IN(32), T0);
    ln_k<<<NT_, 128, 0, stream>>>(T0, IN(33), IN(34), nullptr, T1, 0);
    launch_gemm(stream, T1, C_, w1h, IN(36), nullptr, nullptr, nullptr,
                T0, C_, 0, NT_, C_, C_, 0, 0, 0, ACT_GELU);
    gap_k<<<(B_ * C_ + 255) / 256, 256, 0, stream>>>(XT, GAP);
    launch_gemm(stream, GAP, C_, gw1h, IN(38), nullptr, nullptr, nullptr,
                MID, 96, 0, B_, C_, 96, 0, 0, 0, ACT_GELU);
    launch_gemm(stream, MID, 96, gw2h, IN(40), nullptr, nullptr, nullptr,
                GATE, C_, 0, B_, 96, C_, 0, 0, 0, ACT_SIGM);
    bmul_k<<<EWG, 256, 0, stream>>>(T0, GATE, T1);
    launch_gemm(stream, T1, C_, wfh, IN(42), nullptr, nullptr, XT,
                T2, C_, 0, NT_, C_, C_, 0, 0, 0, ACT_NONE);
    combine_k<<<EWG, 256, 0, stream>>>(ACC, T2, CW, 2);
  }

  // ---- expert 3: frequency ----
  freq_k<<<EWG, 256, 0, stream>>>(XT, T0, T1, T2);
  run_se_fuse(stream, d_in, 43, T0, T1, T2, XT, XT, CAT, GAP, MID, GATE, T3, &wh);
  combine_k<<<EWG, 256, 0, stream>>>(ACC, T3, CW, 3);

  // ---- expert 4: edge ----
  edge_k<<<EWG, 256, 0, stream>>>(XT, T0, T1, T2, T3);
  run_se_fuse(stream, d_in, 67, T0, T1, T2, T3, XT, CAT, GAP, MID, GATE, T0, &wh);
  combine_k<<<EWG, 256, 0, stream>>>(ACC, T0, CW, 4);

  // ---- final: token-major -> NCHW ----
  tok2nchw_k<<<EWG, 256, 0, stream>>>(ACC, (float*)d_out);
}
